// ClusterAttention_78314433675641
// MI455X (gfx1250) — matrix operand order repaired and checked by reading, not compiled, not run on hardware
//
#include <hip/hip_runtime.h>
#include <hip/hip_bf16.h>

// ---------------------------------------------------------------------------
// ClusterAttention for MI455X (gfx1250, wave32, WMMA + async-to-LDS gather)
//   B=4 N=8192 C=384 H=12 D=2 K=32 M=256 ch=16
// All four matmuls use v_wmma_f32_16x16x32_bf16 (f32 accumulate).
// ---------------------------------------------------------------------------

typedef __attribute__((ext_vector_type(16))) __bf16          v16bf;
typedef __attribute__((ext_vector_type(16))) unsigned short  v16u;
typedef __attribute__((ext_vector_type(8)))  float           v8f;

#define Bc   4
#define Nn   8192
#define Cc   384
#define Hh   12
#define Kc   32
#define Mm   256

static __device__ __forceinline__ unsigned short f2bf(float f) {
  unsigned u = __builtin_bit_cast(unsigned, f);
  u += 0x7fffu + ((u >> 16) & 1u);          // round-to-nearest-even
  return (unsigned short)(u >> 16);
}

// 16-bit A-operand (16x32) element -> K index (ISA 7.12.2 layout)
static __device__ __forceinline__ int a_kidx(int e, int kgrp) {
  int p = e >> 1, j = e & 1;
  return (p < 4) ? (kgrp * 8 + 2 * p + j) : (16 + kgrp * 8 + 2 * (p - 4) + j);
}

static __device__ __forceinline__ v8f wmma_bf16(v16u a, v16u b, v8f c) {
  return __builtin_amdgcn_wmma_f32_16x16x32_bf16(
      false, __builtin_bit_cast(v16bf, a),
      false, __builtin_bit_cast(v16bf, b),
      (short)0, c, false, false);
}

// ---------------------------------------------------------------------------
// Kernel 0a: packed f32 -> bf16 conversion (2 elements / thread)
// ---------------------------------------------------------------------------
__global__ __launch_bounds__(256) void cvt_bf16_kernel(const float* __restrict__ src,
                                                       unsigned* __restrict__ dst,
                                                       int npairs) {
  const int i = blockIdx.x * 256 + threadIdx.x;
  if (i < npairs) {
    const float2 f = ((const float2*)src)[i];
    dst[i] = (unsigned)f2bf(f.x) | ((unsigned)f2bf(f.y) << 16);
  }
}

// ---------------------------------------------------------------------------
// Kernel 0b: per-dim max of pos over all B*N rows
// ---------------------------------------------------------------------------
__global__ __launch_bounds__(256) void posmax_kernel(const float* __restrict__ pos,
                                                     float* __restrict__ pmax) {
  __shared__ float r0s[256], r1s[256];
  int tid = threadIdx.x;
  float m0 = -3.0e38f, m1 = -3.0e38f;
  for (int i = tid; i < Bc * Nn; i += 256) {
    m0 = fmaxf(m0, pos[2 * i + 0]);
    m1 = fmaxf(m1, pos[2 * i + 1]);
  }
  r0s[tid] = m0; r1s[tid] = m1;
  __syncthreads();
  for (int s = 128; s > 0; s >>= 1) {
    if (tid < s) {
      r0s[tid] = fmaxf(r0s[tid], r0s[tid + s]);
      r1s[tid] = fmaxf(r1s[tid], r1s[tid + s]);
    }
    __syncthreads();
  }
  if (tid == 0) { pmax[0] = r0s[0]; pmax[1] = r1s[0]; }
}

// ---------------------------------------------------------------------------
// Kernel 1: qkv = feat @ w_qkv^T + b_qkv  -> bf16, layout (b, h, n, 96)
// Each wave: one 16-row x 96-col strip (= one head's channels), A fragment
// reused across 6 wmma per K-step; K = 384 in 12 steps. Pure bf16 loads.
// ---------------------------------------------------------------------------
__global__ __launch_bounds__(256) void qkv_kernel(const unsigned short* __restrict__ fb,
                                                  const unsigned short* __restrict__ wqb,
                                                  const float* __restrict__ bq,
                                                  unsigned short* __restrict__ qkv) {
  const int wt      = blockIdx.x * 8 + (threadIdx.x >> 5);   // 2048*12 strips
  const int rowTile = wt / Hh;
  const int h       = wt % Hh;
  const int r0 = rowTile * 16;
  const int lane = threadIdx.x & 31, nl = lane & 15, kgrp = lane >> 4;

  const unsigned short* arow = fb + (size_t)(r0 + nl) * Cc;

  v8f acc[6];
#pragma unroll
  for (int t = 0; t < 6; ++t) {
    v8f z = {0.f, 0.f, 0.f, 0.f, 0.f, 0.f, 0.f, 0.f};
    acc[t] = z;
  }

  for (int ks = 0; ks < 12; ++ks) {
    const int k0 = ks * 32;
    __builtin_prefetch(arow + k0 + 32, 0, 1);
    v16u a;
#pragma unroll
    for (int e = 0; e < 16; ++e) a[e] = arow[k0 + kgrp * 16 + e];
#pragma unroll
    for (int t = 0; t < 6; ++t) {
      v16u b;
#pragma unroll
      for (int e = 0; e < 16; ++e)
        b[e] = wqb[(size_t)(h * 96 + t * 16 + nl) * Cc + k0 + kgrp * 16 + e];
      acc[t] = wmma_bf16(a, b, acc[t]);
    }
  }

#pragma unroll
  for (int t = 0; t < 6; ++t) {
    const int   cc   = h * 96 + t * 16 + nl;
    const float bias = bq[cc];
    const int   rem  = t * 16 + nl;
#pragma unroll
    for (int e = 0; e < 8; ++e) {
      const int r  = r0 + e + kgrp * 8;       // D-matrix row layout
      const int bI = r >> 13, nn = r & (Nn - 1);
      qkv[(((size_t)bI * Hh + h) * Nn + nn) * 96 + rem] = f2bf(acc[t][e] + bias);
    }
  }
}

// ---------------------------------------------------------------------------
// Kernel 2: fused clustered attention.
// One block per (cluster kc in [0,128), head h).
//  - q/k gathered via global_load_async_to_lds_b128 into K-padded LDS rows
//    (32 ushorts/row, upper half zeroed -> fragment loads are contiguous
//    ds_load_b128, no zero-pad selects).
//  - v gathered through registers and stored TRANSPOSED (sh_vt[c][j]) so the
//    P x V B-fragment is 2 contiguous ds_load_b128 instead of 16 scalars.
//  - 256x256 logits via WMMA, register softmax (16-lane shfl_xor), P staged
//    through LDS (s_wait_dscnt) to reach the A-operand layout, 4 PV WMMAs per
//    32-col block, scatter to feat2 (b, n, h*64+c) bf16.
// ---------------------------------------------------------------------------
__global__ __launch_bounds__(256) void attn_kernel(const unsigned short* __restrict__ qkv,
                                                   const int*   __restrict__ midx,
                                                   const float* __restrict__ pos,
                                                   const float* __restrict__ w_pos,
                                                   const float* __restrict__ b_pos,
                                                   const float* __restrict__ pmax,
                                                   unsigned short* __restrict__ feat2) {
  __shared__ unsigned short sh_q[Mm * 32];     // padded: K 16..31 zeroed
  __shared__ unsigned short sh_k[Mm * 32];     // padded: K 16..31 zeroed
  __shared__ unsigned short sh_vt[64 * Mm];    // transposed: [c][j]
  __shared__ float          sh_s[Mm];
  __shared__ int            sh_n[Mm];
  __shared__ unsigned short sh_p[8][16 * 32];

  const int h    = blockIdx.x % Hh;
  const int kc   = blockIdx.x / Hh;      // (b, k-cluster)
  const int b    = kc / Kc;
  const int kcl  = kc % Kc;
  const int row  = b * Hh + h;           // row into qkv / member_idx
  const int tid  = threadIdx.x;
  const int lane = tid & 31;
  const int wave = tid >> 5;
  const int nl   = lane & 15;
  const int kgrp = lane >> 4;

  // ---- gather: 256 member rows (192B each) into LDS ---------------------
  {
    const int i = tid;
    const int n = midx[((size_t)row * Kc + kcl) * Mm + i];
    sh_n[i] = n;
    const unsigned short* srow = qkv + ((size_t)row * Nn + n) * 96;
    const unsigned long long gq = (unsigned long long)(uintptr_t)srow;   // bytes 0..31  (q)
    const unsigned long long gk = gq + 32;                               // bytes 32..63 (k)
    const unsigned lq = (unsigned)(uintptr_t)(sh_q + (size_t)i * 32);
    const unsigned lk = (unsigned)(uintptr_t)(sh_k + (size_t)i * 32);
#pragma unroll
    for (int t = 0; t < 2; ++t)
      __asm__ volatile("global_load_async_to_lds_b128 %0, %1, off offset:%c2"
                       :: "v"(lq), "v"(gq), "i"(t * 16) : "memory");
#pragma unroll
    for (int t = 0; t < 2; ++t)
      __asm__ volatile("global_load_async_to_lds_b128 %0, %1, off offset:%c2"
                       :: "v"(lk), "v"(gk), "i"(t * 16) : "memory");
    // zero the K=16..31 padding (bytes 32..63 of each row; disjoint from async)
    const uint4 z4 = {0u, 0u, 0u, 0u};
    ((uint4*)(sh_q + (size_t)i * 32 + 16))[0] = z4;
    ((uint4*)(sh_q + (size_t)i * 32 + 24))[0] = z4;
    ((uint4*)(sh_k + (size_t)i * 32 + 16))[0] = z4;
    ((uint4*)(sh_k + (size_t)i * 32 + 24))[0] = z4;

    // v: load row (64 bf16) through registers, store transposed
    const unsigned* srcv = (const unsigned*)(srow + 32);
#pragma unroll
    for (int w = 0; w < 32; ++w) {
      const unsigned d = srcv[w];
      sh_vt[(2 * w + 0) * Mm + i] = (unsigned short)(d & 0xffffu);
      sh_vt[(2 * w + 1) * Mm + i] = (unsigned short)(d >> 16);
    }

    const float p0 = pos[((size_t)b * Nn + n) * 2 + 0] / pmax[0];
    const float p1 = pos[((size_t)b * Nn + n) * 2 + 1] / pmax[1];
    sh_s[i] = p0 * w_pos[h * 2 + 0] + p1 * w_pos[h * 2 + 1];
  }
  __asm__ volatile("s_wait_asynccnt 0" ::: "memory");   // drain async copies
  __syncthreads();

  const float scale = 0.17677669529663687f;   // (C/H)^-0.5 = 1/sqrt(32)
  const float bp    = b_pos[h];

  for (int tt = 0; tt < 2; ++tt) {            // each wave: two 16-row strips
    const int r0 = wave * 32 + tt * 16;

    // Q strip as A operand (padding already zeroed in LDS -> contiguous)
    v16u aq;
#pragma unroll
    for (int e = 0; e < 16; ++e)
      aq[e] = sh_q[(r0 + nl) * 32 + kgrp * 16 + e];
    float s_i[8];
#pragma unroll
    for (int e = 0; e < 8; ++e) s_i[e] = sh_s[r0 + e + kgrp * 8];

    // ---- logits: 16 column tiles of QK^T -------------------------------
    v8f S[16];
#pragma unroll
    for (int jt = 0; jt < 16; ++jt) {
      v16u bk;
#pragma unroll
      for (int e = 0; e < 16; ++e)
        bk[e] = sh_k[(jt * 16 + nl) * 32 + kgrp * 16 + e];
      v8f z = {0.f, 0.f, 0.f, 0.f, 0.f, 0.f, 0.f, 0.f};
      S[jt] = wmma_bf16(aq, bk, z);
    }

    // ---- bias + row max -------------------------------------------------
    float lmax[8];
#pragma unroll
    for (int e = 0; e < 8; ++e) lmax[e] = -3.0e38f;
#pragma unroll
    for (int jt = 0; jt < 16; ++jt) {
      const float s_j = sh_s[jt * 16 + nl];
#pragma unroll
      for (int e = 0; e < 8; ++e) {
        const float l = scale * S[jt][e] + s_j - s_i[e] + bp;
        S[jt][e] = l;
        lmax[e] = fmaxf(lmax[e], l);
      }
    }
#pragma unroll
    for (int e = 0; e < 8; ++e)
#pragma unroll
      for (int msk = 1; msk < 16; msk <<= 1)
        lmax[e] = fmaxf(lmax[e], __shfl_xor(lmax[e], msk, 32));

    // ---- exp, row sum, P x V -------------------------------------------
    float lsum[8];
#pragma unroll
    for (int e = 0; e < 8; ++e) lsum[e] = 0.f;
    v8f O[4];
#pragma unroll
    for (int ct = 0; ct < 4; ++ct) {
      v8f z = {0.f, 0.f, 0.f, 0.f, 0.f, 0.f, 0.f, 0.f};
      O[ct] = z;
    }

#pragma unroll
    for (int jt = 0; jt < 16; ++jt) {
#pragma unroll
      for (int e = 0; e < 8; ++e) {
        const float ev = __expf(S[jt][e] - lmax[e]);
        lsum[e] += ev;
        sh_p[wave][(e + kgrp * 8) * 32 + (jt & 1) * 16 + nl] = f2bf(ev);
      }
      if (jt & 1) {
        // wave-private LDS RAW: drain the DS split counter (CDNA5)
        __asm__ volatile("s_wait_dscnt 0" ::: "memory");
        v16u ap;
#pragma unroll
        for (int e = 0; e < 16; ++e)
          ap[e] = sh_p[wave][nl * 32 + kgrp * 16 + e];
        const int j0 = (jt - 1) * 16;
#pragma unroll
        for (int ct = 0; ct < 4; ++ct) {
          v16u bv;
#pragma unroll
          for (int e = 0; e < 16; ++e)
            bv[e] = sh_vt[(ct * 16 + nl) * Mm + j0 + kgrp * 16 + e];
          O[ct] = wmma_bf16(ap, bv, O[ct]);
        }
      }
    }

#pragma unroll
    for (int e = 0; e < 8; ++e) {
#pragma unroll
      for (int msk = 1; msk < 16; msk <<= 1)
        lsum[e] += __shfl_xor(lsum[e], msk, 32);
      lsum[e] = 1.f / lsum[e];
    }

    // ---- normalize + scatter into proj-input layout --------------------
#pragma unroll
    for (int ct = 0; ct < 4; ++ct) {
#pragma unroll
      for (int e = 0; e < 8; ++e) {
        const float val = O[ct][e] * lsum[e];
        const int i = r0 + e + kgrp * 8;
        const int n = sh_n[i];
        feat2[(((size_t)b * Nn + n) * Hh + h) * 64 + ct * 16 + nl] = f2bf(val);
      }
    }
  }
}

// ---------------------------------------------------------------------------
// Kernel 3: out = feat2 @ w_proj^T + b_proj   (f32 output)
// Each wave: 16-row x 96-col strip, A fragment reused across 6 wmma/K-step.
// ---------------------------------------------------------------------------
__global__ __launch_bounds__(256) void proj_kernel(const unsigned short* __restrict__ feat2,
                                                   const unsigned short* __restrict__ wpb,
                                                   const float* __restrict__ bpj,
                                                   float* __restrict__ out) {
  const int wt      = blockIdx.x * 8 + (threadIdx.x >> 5);   // 2048*4 strips
  const int rowTile = wt >> 2;
  const int g       = wt & 3;                                // 4 groups of 96 cols
  const int r0 = rowTile * 16, c0 = g * 96;
  const int lane = threadIdx.x & 31, nl = lane & 15, kgrp = lane >> 4;

  const unsigned short* arow = feat2 + (size_t)(r0 + nl) * (2 * Cc);

  v8f acc[6];
#pragma unroll
  for (int t = 0; t < 6; ++t) {
    v8f z = {0.f, 0.f, 0.f, 0.f, 0.f, 0.f, 0.f, 0.f};
    acc[t] = z;
  }

  for (int ks = 0; ks < 24; ++ks) {
    const int k0 = ks * 32;
    __builtin_prefetch(arow + k0 + 32, 0, 1);
    v16u a;
#pragma unroll
    for (int e = 0; e < 16; ++e) a[e] = arow[k0 + kgrp * 16 + e];
#pragma unroll
    for (int t = 0; t < 6; ++t) {
      v16u b;
#pragma unroll
      for (int e = 0; e < 16; ++e)
        b[e] = wpb[(size_t)(c0 + t * 16 + nl) * (2 * Cc) + k0 + kgrp * 16 + e];
      acc[t] = wmma_bf16(a, b, acc[t]);
    }
  }

#pragma unroll
  for (int t = 0; t < 6; ++t) {
    const int   cc   = c0 + t * 16 + nl;
    const float bias = bpj[cc];
#pragma unroll
    for (int e = 0; e < 8; ++e) {
      const int r = r0 + e + kgrp * 8;
      out[(size_t)r * Cc + cc] = acc[t][e] + bias;
    }
  }
}

// ---------------------------------------------------------------------------
extern "C" void kernel_launch(void* const* d_in, const int* in_sizes, int n_in,
                              void* d_out, int out_size, void* d_ws, size_t ws_size,
                              hipStream_t stream) {
  const float* pos    = (const float*)d_in[0];
  const float* feat   = (const float*)d_in[1];
  const int*   midx   = (const int*)  d_in[2];
  const float* w_qkv  = (const float*)d_in[3];
  const float* b_qkv  = (const float*)d_in[4];
  const float* w_pos  = (const float*)d_in[5];
  const float* b_pos  = (const float*)d_in[6];
  const float* w_proj = (const float*)d_in[7];
  const float* b_proj = (const float*)d_in[8];
  float* out = (float*)d_out;

  // workspace layout (bytes):
  //   [  0,  256) pmax
  //   qkv   bf16 (48,8192,96)   = 75,497,472
  //   feat2 bf16 (4,8192,768)   = 50,331,648
  //   featb bf16 (32768,384)    = 25,165,824
  //   wqb   bf16 (1152,384)     =    884,736
  //   wpb   bf16 (384,768)      =    589,824
  float* pmax = (float*)d_ws;
  unsigned short* qkv   = (unsigned short*)((char*)d_ws + 256);
  unsigned short* feat2 = qkv   + (size_t)Bc * Hh * Nn * 96;
  unsigned short* featb = feat2 + (size_t)Bc * Nn * 2 * Cc;
  unsigned short* wqb   = featb + (size_t)Bc * Nn * Cc;
  unsigned short* wpb   = wqb   + (size_t)3 * Cc * Cc;

  posmax_kernel<<<1, 256, 0, stream>>>(pos, pmax);
  // pre-convert f32 -> bf16 (packed pairs)
  cvt_bf16_kernel<<<(Bc * Nn * Cc / 2 + 255) / 256, 256, 0, stream>>>(feat,   (unsigned*)featb, Bc * Nn * Cc / 2);
  cvt_bf16_kernel<<<(3 * Cc * Cc / 2 + 255) / 256, 256, 0, stream>>>(w_qkv,  (unsigned*)wqb,   3 * Cc * Cc / 2);
  cvt_bf16_kernel<<<(Cc * 2 * Cc / 2 + 255) / 256, 256, 0, stream>>>(w_proj, (unsigned*)wpb,   Cc * 2 * Cc / 2);
  // 2048 row tiles * 12 head-strips / 8 waves per block
  qkv_kernel<<<3072, 256, 0, stream>>>(featb, wqb, b_qkv, qkv);
  // one block per (cluster, head): 128 * 12
  attn_kernel<<<1536, 256, 0, stream>>>(qkv, midx, pos, w_pos, b_pos, pmax, feat2);
  // 2048 row tiles * 4 col-groups / 8 waves per block
  proj_kernel<<<1024, 256, 0, stream>>>(feat2, wpb, b_proj, out);
}
